// WanI2VCrossAttention_72550587564137
// MI455X (gfx1250) — matrix operand order repaired and checked by reading, not compiled, not run on hardware
//
#include <hip/hip_runtime.h>
#include <hip/hip_bf16.h>
#include <stdint.h>

#define DIMV  2048
#define HEADS 16
#define HD    128
#define SEQ   4096
#define BATCH 2
#define NCTX  769
#define NIMG  257
#define NTXT  512
#define LPAD_T 512            // txt keys padded (already multiple of 32)
#define LPAD_I 288            // img keys 257 -> 288
#define EPSV  1e-6f

typedef __attribute__((ext_vector_type(16))) __bf16 bf16x16;
typedef __attribute__((ext_vector_type(8)))  float  f32x8;
typedef int v4i __attribute__((vector_size(16)));

static __device__ __forceinline__ unsigned short f2bf(float f) {
  unsigned int u = __builtin_bit_cast(unsigned int, f);
  unsigned int r = u + 0x7FFFu + ((u >> 16) & 1u);
  return (unsigned short)(r >> 16);
}
static __device__ __forceinline__ __bf16 bfbits(unsigned short u) {
  return __builtin_bit_cast(__bf16, u);
}
static __device__ __forceinline__ f32x8 wmma_bf16(bf16x16 a, bf16x16 b, f32x8 c) {
  return __builtin_amdgcn_wmma_f32_16x16x32_bf16(false, a, false, b, (short)0, c,
                                                 false, false);
}

// ---------------------------------------------------------------------------
// Async global->LDS staging (CDNA5 ASYNCcnt path) with synchronous fallback.
// Probe-confirmed: builtin exists, signature is (global v4i*, lds v4i*, imm, imm).
// ---------------------------------------------------------------------------
#if __has_builtin(__builtin_amdgcn_global_load_async_to_lds_b128)
#define HAVE_ASYNC 1
#else
#define HAVE_ASYNC 0
#endif

static __device__ __forceinline__ void stage16(const unsigned short* g,
                                               unsigned short* l) {
#if HAVE_ASYNC
  __builtin_amdgcn_global_load_async_to_lds_b128(
      (__attribute__((address_space(1))) v4i*)(v4i*)g,
      (__attribute__((address_space(3))) v4i*)(v4i*)l, 0, 0);
#else
  *(uint4*)l = *(const uint4*)g;   // all call sites are 16B aligned
#endif
}
static __device__ __forceinline__ void stage_wait() {
#if HAVE_ASYNC
#if __has_builtin(__builtin_amdgcn_s_wait_asynccnt)
  __builtin_amdgcn_s_wait_asynccnt(0);
#else
  asm volatile("s_wait_asynccnt 0x0" ::: "memory");
#endif
#endif
}

// ---------------------------------------------------------------------------
// fp32 -> bf16 elementwise convert
// ---------------------------------------------------------------------------
__global__ void k_cvt_bf16(const float* __restrict__ src,
                           unsigned short* __restrict__ dst, long n) {
  long i = ((long)blockIdx.x * blockDim.x + threadIdx.x) * 4;
  if (i + 3 < n) {
    float4 v = *(const float4*)(src + i);
    ushort4 o;
    o.x = f2bf(v.x); o.y = f2bf(v.y); o.z = f2bf(v.z); o.w = f2bf(v.w);
    *(ushort4*)(dst + i) = o;
  } else {
    for (long j = i; j < n && j >= 0; ++j) dst[j] = f2bf(src[j]);
  }
}

// fp32 W[k][n] -> bf16 W^T[n][k]   (DIMV x DIMV)
__global__ void k_cvt_wT(const float* __restrict__ w,
                         unsigned short* __restrict__ wt) {
  long idx = ((long)blockIdx.x * blockDim.x + threadIdx.x) * 4;
  int k = (int)(idx >> 11);
  int n = (int)(idx & 2047);
  float4 v = *(const float4*)(w + idx);
  wt[(long)(n + 0) * DIMV + k] = f2bf(v.x);
  wt[(long)(n + 1) * DIMV + k] = f2bf(v.y);
  wt[(long)(n + 2) * DIMV + k] = f2bf(v.z);
  wt[(long)(n + 3) * DIMV + k] = f2bf(v.w);
}

// fp32 V[(b*L+l)][h*128+d] -> bf16 vT[((b*H+h)*128+d)*Lpad + l], zero padding
__global__ void k_cvt_vT(const float* __restrict__ vf,
                         unsigned short* __restrict__ vT, int L, int Lpad) {
  long idx = (long)blockIdx.x * blockDim.x + threadIdx.x;
  long total = (long)BATCH * Lpad * DIMV;
  if (idx >= total) return;
  int c = (int)(idx % DIMV);
  long t = idx / DIMV;
  int b = (int)(t / Lpad);
  int l = (int)(t % Lpad);
  int h = c >> 7, d = c & 127;
  unsigned short v = 0;
  if (l < L) v = f2bf(vf[((long)(b * L + l)) * DIMV + c]);
  vT[((long)((b * HEADS + h) * HD + d)) * Lpad + l] = v;
}

// ---------------------------------------------------------------------------
// RMS norm over a DIMV row * scale * extra -> bf16.  Rows are (b, l<L) in the
// f32 input; output row index uses padded per-batch stride Lpad.
// ---------------------------------------------------------------------------
__global__ __launch_bounds__(256) void k_rms_bf16(
    const float* __restrict__ y, const float* __restrict__ scale,
    unsigned short* __restrict__ out, float extra, int L, int Lpad) {
  __shared__ float red[256];
  const long row = blockIdx.x;
  const long orow = (row / L) * (long)Lpad + (row % L);
  const float* yr = y + row * (long)DIMV;
  const int base = threadIdx.x * 8;
  float vals[8];
  float ss = 0.f;
#pragma unroll
  for (int j = 0; j < 8; j += 4) {
    float4 v = *(const float4*)(yr + base + j);
    vals[j] = v.x; vals[j + 1] = v.y; vals[j + 2] = v.z; vals[j + 3] = v.w;
    ss += v.x * v.x + v.y * v.y + v.z * v.z + v.w * v.w;
  }
  red[threadIdx.x] = ss;
  __syncthreads();
  for (int s = 128; s > 0; s >>= 1) {
    if ((int)threadIdx.x < s) red[threadIdx.x] += red[threadIdx.x + s];
    __syncthreads();
  }
  const float inv = rsqrtf(red[0] / (float)DIMV + EPSV) * extra;
#pragma unroll
  for (int j = 0; j < 8; ++j) {
    int c = base + j;
    out[orow * (long)DIMV + c] = f2bf(vals[j] * inv * scale[c]);
  }
}

// ---------------------------------------------------------------------------
// bf16 GEMM: C(MxN,f32) = A(MxK) * W^T(NxK)^T + bias
// Block tile 128x128, K-step 32, 8 waves, wave tile 32x64 (8 WMMA/K-step).
// Double-buffered LDS, async global->LDS staging.
// ---------------------------------------------------------------------------
#define BM 128
#define BN 128
#define BKS 32
#define AS_STRIDE 40
#define BT_STRIDE 40

__global__ __launch_bounds__(256) void k_gemm_bf16(
    const unsigned short* __restrict__ A, const unsigned short* __restrict__ WT,
    const float* __restrict__ bias, float* __restrict__ C, int M, int N, int K) {
  __shared__ unsigned short As[2][BM * AS_STRIDE];
  __shared__ unsigned short Bt[2][BN * BT_STRIDE];

  const int tid = threadIdx.x;
  const int lane = tid & 31;
  const int wid = tid >> 5;
  const int m0 = blockIdx.y * BM;
  const int n0 = blockIdx.x * BN;
  const int waveM = (wid & 3) * 32;
  const int waveN = (wid >> 2) * 64;
  const int lm = lane & 15;
  const int lh = lane >> 4;

  f32x8 acc[2][4];
  const f32x8 zero = {0.f, 0.f, 0.f, 0.f, 0.f, 0.f, 0.f, 0.f};
#pragma unroll
  for (int mt = 0; mt < 2; ++mt)
#pragma unroll
    for (int nt = 0; nt < 4; ++nt) acc[mt][nt] = zero;

  auto stage = [&](int buf, int kc) {
#pragma unroll
    for (int it = 0; it < 2; ++it) {
      int li = tid + it * 256;        // 0..511
      int r = li >> 2;                // 0..127
      int cb = (li & 3) * 8;          // 0..24
      stage16(A + (long)(m0 + r) * K + kc + cb, &As[buf][r * AS_STRIDE + cb]);
      stage16(WT + (long)(n0 + r) * K + kc + cb, &Bt[buf][r * BT_STRIDE + cb]);
    }
  };

  stage(0, 0);
  stage_wait();
  __syncthreads();

  int cur = 0;
  for (int kc = 0; kc < K; kc += BKS) {
    const int nxt = kc + BKS;
    if (nxt < K) stage(cur ^ 1, nxt);

    // A fragments (two 16x32 tiles, ISA layout)
    bf16x16 af[2];
#pragma unroll
    for (int mt = 0; mt < 2; ++mt) {
#pragma unroll
      for (int v = 0; v < 8; ++v) {
        int k = ((v < 4) ? 0 : 16) + lh * 8 + (v & 3) * 2;
        unsigned int w = *(const unsigned int*)(
            &As[cur][(waveM + mt * 16 + lm) * AS_STRIDE + k]);
        af[mt][2 * v]     = bfbits((unsigned short)(w & 0xFFFFu));
        af[mt][2 * v + 1] = bfbits((unsigned short)(w >> 16));
      }
    }
#pragma unroll
    for (int nt = 0; nt < 4; ++nt) {
      int ncol = waveN + nt * 16 + lm;
      bf16x16 bfr;
#pragma unroll
      for (int v = 0; v < 8; ++v) {
        int k = ((v < 4) ? 0 : 16) + lh * 8 + (v & 3) * 2;
        unsigned int w =
            *(const unsigned int*)(&Bt[cur][ncol * BT_STRIDE + k]);
        bfr[2 * v]     = bfbits((unsigned short)(w & 0xFFFFu));
        bfr[2 * v + 1] = bfbits((unsigned short)(w >> 16));
      }
      acc[0][nt] = wmma_bf16(af[0], bfr, acc[0][nt]);
      acc[1][nt] = wmma_bf16(af[1], bfr, acc[1][nt]);
    }

    if (nxt < K) stage_wait();
    __syncthreads();
    cur ^= 1;
  }

#pragma unroll
  for (int nt = 0; nt < 4; ++nt) {
    int gc = n0 + waveN + nt * 16 + lm;
    float bv = bias ? bias[gc] : 0.f;
#pragma unroll
    for (int mt = 0; mt < 2; ++mt) {
#pragma unroll
      for (int r = 0; r < 8; ++r) {
        int gr = m0 + waveM + mt * 16 + r + lh * 8;
        if (gr < M) C[(long)gr * N + gc] = acc[mt][nt][r] + bv;
      }
    }
  }
}

// ---------------------------------------------------------------------------
// Flash attention, bf16 WMMA, double-buffered async K/V staging.
// K: (b, keyPad, 2048) bf16.  V: pre-transposed (b, h, d, keyPad) bf16.
// mode 0: Of32 = result (txt).  mode 1: Obf16 = bf16(Of32 + result) (img).
// ---------------------------------------------------------------------------
#define KT 32
#define KS_STRIDE 136
#define VT_STRIDE 40
#define PS_STRIDE 36

__global__ __launch_bounds__(256) void k_attn(
    const unsigned short* __restrict__ Q, const unsigned short* __restrict__ Kb,
    const unsigned short* __restrict__ vT, float* __restrict__ Of32,
    unsigned short* __restrict__ Obf16, int S, int L, int Lpad, int mode) {
  __shared__ unsigned short Ks[2][KT * KS_STRIDE];
  __shared__ unsigned short Vt[2][HD * VT_STRIDE];
  __shared__ unsigned short Ps[8 * 16 * PS_STRIDE];

  const int tid = threadIdx.x, lane = tid & 31, wid = tid >> 5;
  const int b = blockIdx.z, h = blockIdx.y;
  const int q0 = blockIdx.x * 128 + wid * 16;
  const int hc = h * HD;
  const int lm = lane & 15, lh = lane >> 4;
  unsigned short* Pw = &Ps[wid * 16 * PS_STRIDE];
  const f32x8 zero = {0.f, 0.f, 0.f, 0.f, 0.f, 0.f, 0.f, 0.f};
  const unsigned short* vTh = vT + ((long)(b * HEADS + h) * HD) * Lpad;

  auto stageKV = [&](int buf, int k0) {
#pragma unroll
    for (int it = 0; it < 2; ++it) {
      int li = tid + it * 256;          // 0..511
      int kk = li >> 4, cb = (li & 15) * 8;
      stage16(Kb + ((long)(b * Lpad + k0 + kk)) * DIMV + hc + cb,
              &Ks[buf][kk * KS_STRIDE + cb]);
      int d = li >> 2, j = (li & 3) * 8;
      stage16(vTh + (long)d * Lpad + k0 + j, &Vt[buf][d * VT_STRIDE + j]);
    }
  };

  // Preload Q as 4 A-fragments (16x32 each) straight from global
  bf16x16 qf[4];
  const unsigned short* qrow = Q + ((long)(b * S + q0 + lm)) * DIMV + hc;
#pragma unroll
  for (int c = 0; c < 4; ++c) {
#pragma unroll
    for (int v = 0; v < 8; ++v) {
      int k = c * 32 + ((v < 4) ? 0 : 16) + lh * 8 + (v & 3) * 2;
      unsigned int w = *(const unsigned int*)(qrow + k);
      qf[c][2 * v]     = bfbits((unsigned short)(w & 0xFFFFu));
      qf[c][2 * v + 1] = bfbits((unsigned short)(w >> 16));
    }
  }

  float mrow[8], lrow[8];
  f32x8 o[8];
#pragma unroll
  for (int r = 0; r < 8; ++r) { mrow[r] = -3.0e38f; lrow[r] = 0.f; }
#pragma unroll
  for (int t = 0; t < 8; ++t) o[t] = zero;

  stageKV(0, 0);
  stage_wait();
  __syncthreads();

  int cur = 0;
  for (int k0 = 0; k0 < L; k0 += KT) {
    const int nxt = k0 + KT;
    if (nxt < L) stageKV(cur ^ 1, nxt);

    // S = Q * K^T : two 16x16 f32 tiles (4 WMMAs each over the 128 dim)
    f32x8 s[2];
    s[0] = zero; s[1] = zero;
#pragma unroll
    for (int nt = 0; nt < 2; ++nt) {
      int key = nt * 16 + lm;
#pragma unroll
      for (int c = 0; c < 4; ++c) {
        bf16x16 bfr;
#pragma unroll
        for (int v = 0; v < 8; ++v) {
          int d = c * 32 + ((v < 4) ? 0 : 16) + lh * 8 + (v & 3) * 2;
          unsigned int w =
              *(const unsigned int*)(&Ks[cur][key * KS_STRIDE + d]);
          bfr[2 * v]     = bfbits((unsigned short)(w & 0xFFFFu));
          bfr[2 * v + 1] = bfbits((unsigned short)(w >> 16));
        }
        s[nt] = wmma_bf16(qf[c], bfr, s[nt]);
      }
      if (k0 + nt * 16 + lm >= L) {   // mask padded keys (column == lane)
#pragma unroll
        for (int r = 0; r < 8; ++r) s[nt][r] = -3.0e38f;
      }
    }

    // Online softmax (row reductions inside each 16-lane half)
#pragma unroll
    for (int r = 0; r < 8; ++r) {
      float mx = fmaxf(s[0][r], s[1][r]);
#pragma unroll
      for (int sm = 8; sm >= 1; sm >>= 1) mx = fmaxf(mx, __shfl_xor(mx, sm, 32));
      float mnew = fmaxf(mrow[r], mx);
      float alpha = __expf(mrow[r] - mnew);
      float p0 = __expf(s[0][r] - mnew);
      float p1 = __expf(s[1][r] - mnew);
      float ps = p0 + p1;
#pragma unroll
      for (int sm = 8; sm >= 1; sm >>= 1) ps += __shfl_xor(ps, sm, 32);
      lrow[r] = lrow[r] * alpha + ps;
      mrow[r] = mnew;
#pragma unroll
      for (int t = 0; t < 8; ++t) o[t][r] *= alpha;
      int prow = r + lh * 8;
      Pw[prow * PS_STRIDE + lm]      = f2bf(p0);
      Pw[prow * PS_STRIDE + 16 + lm] = f2bf(p1);
    }

    // Re-load P as A-fragment (C-layout -> A-layout via wave-private LDS)
    bf16x16 pf;
#pragma unroll
    for (int v = 0; v < 8; ++v) {
      int k = ((v < 4) ? 0 : 16) + lh * 8 + (v & 3) * 2;
      unsigned int w = *(const unsigned int*)(&Pw[lm * PS_STRIDE + k]);
      pf[2 * v]     = bfbits((unsigned short)(w & 0xFFFFu));
      pf[2 * v + 1] = bfbits((unsigned short)(w >> 16));
    }
    // O += P * V
#pragma unroll
    for (int dt = 0; dt < 8; ++dt) {
      int d = dt * 16 + lm;
      bf16x16 bfr;
#pragma unroll
      for (int v = 0; v < 8; ++v) {
        int kk = ((v < 4) ? 0 : 16) + lh * 8 + (v & 3) * 2;
        unsigned int w = *(const unsigned int*)(&Vt[cur][d * VT_STRIDE + kk]);
        bfr[2 * v]     = bfbits((unsigned short)(w & 0xFFFFu));
        bfr[2 * v + 1] = bfbits((unsigned short)(w >> 16));
      }
      o[dt] = wmma_bf16(pf, bfr, o[dt]);
    }

    if (nxt < L) stage_wait();
    __syncthreads();
    cur ^= 1;
  }

  // Epilogue
#pragma unroll
  for (int r = 0; r < 8; ++r) {
    float inv = 1.0f / lrow[r];
    long grow = (long)(b * S + q0 + r + lh * 8);
#pragma unroll
    for (int dt = 0; dt < 8; ++dt) {
      long addr = grow * DIMV + hc + dt * 16 + lm;
      float val = o[dt][r] * inv;
      if (mode == 0) Of32[addr] = val;
      else Obf16[addr] = f2bf(Of32[addr] + val);
    }
  }
}

// ---------------------------------------------------------------------------
// Host-side orchestration
// ---------------------------------------------------------------------------
extern "C" void kernel_launch(void* const* d_in, const int* in_sizes, int n_in,
                              void* d_out, int out_size, void* d_ws,
                              size_t ws_size, hipStream_t stream) {
  (void)in_sizes; (void)n_in; (void)out_size; (void)ws_size;
  const float* x     = (const float*)d_in[0];
  const float* ctx   = (const float*)d_in[1];
  const float* q_w   = (const float*)d_in[2];
  const float* q_b   = (const float*)d_in[3];
  const float* k_w   = (const float*)d_in[4];
  const float* k_b   = (const float*)d_in[5];
  const float* v_w   = (const float*)d_in[6];
  const float* v_b   = (const float*)d_in[7];
  const float* ki_w  = (const float*)d_in[8];
  const float* ki_b  = (const float*)d_in[9];
  const float* vi_w  = (const float*)d_in[10];
  const float* vi_b  = (const float*)d_in[11];
  const float* o_w   = (const float*)d_in[12];
  const float* o_b   = (const float*)d_in[13];
  const float* nq_s  = (const float*)d_in[14];
  const float* nkt_s = (const float*)d_in[15];
  const float* nki_s = (const float*)d_in[16];
  float* out = (float*)d_out;

  char* base = (char*)d_ws;
  size_t off = 0;
  auto alloc = [&](size_t bytes) -> void* {
    void* p = base + off;
    off = (off + bytes + 255) & ~(size_t)255;
    return p;
  };
  const long MQ = (long)BATCH * SEQ;
  const long MT = (long)BATCH * NTXT;
  const long MI = (long)BATCH * NIMG;
  const long W  = (long)DIMV * DIMV;

  unsigned short* wqT  = (unsigned short*)alloc(W * 2);  // W^T bf16
  unsigned short* wkT  = (unsigned short*)alloc(W * 2);
  unsigned short* wvT  = (unsigned short*)alloc(W * 2);
  unsigned short* wkiT = (unsigned short*)alloc(W * 2);
  unsigned short* wviT = (unsigned short*)alloc(W * 2);
  unsigned short* woT  = (unsigned short*)alloc(W * 2);
  unsigned short* xb   = (unsigned short*)alloc(MQ * DIMV * 2);
  unsigned short* cb   = (unsigned short*)alloc((long)BATCH * NCTX * DIMV * 2);
  float* qf  = (float*)alloc(MQ * DIMV * 4);
  float* ktf = (float*)alloc(MT * DIMV * 4);
  float* vtf = (float*)alloc(MT * DIMV * 4);
  float* kif = (float*)alloc(MI * DIMV * 4);
  float* vif = (float*)alloc(MI * DIMV * 4);
  unsigned short* qb16 = (unsigned short*)alloc(MQ * DIMV * 2);
  unsigned short* ktb  = (unsigned short*)alloc((long)BATCH * LPAD_T * DIMV * 2);
  unsigned short* kib  = (unsigned short*)alloc((long)BATCH * LPAD_I * DIMV * 2);
  unsigned short* vTt  = (unsigned short*)alloc((long)BATCH * DIMV * LPAD_T * 2);
  unsigned short* vTi  = (unsigned short*)alloc((long)BATCH * DIMV * LPAD_I * 2);
  float* attnf = qf;            // reuse: q f32 dead after rms-norm pass
  unsigned short* attnb = xb;   // reuse: x bf16 dead after q projection

  auto cvt = [&](const float* s, unsigned short* d, long n) {
    long blocks = (n + 1023) / 1024;
    k_cvt_bf16<<<dim3((unsigned)blocks), 256, 0, stream>>>(s, d, n);
  };
  auto cvtWT = [&](const float* s, unsigned short* d) {
    k_cvt_wT<<<dim3((unsigned)(W / 1024)), 256, 0, stream>>>(s, d);
  };
  auto gemm = [&](const unsigned short* A, const unsigned short* BT,
                  const float* bias, float* C, int M) {
    dim3 grid(DIMV / BN, (M + BM - 1) / BM);
    k_gemm_bf16<<<grid, 256, 0, stream>>>(A, BT, bias, C, M, DIMV, DIMV);
  };

  // 1. fp32 -> bf16 conversions (weights transposed for async-friendly GEMM)
  cvtWT(q_w, wqT);  cvtWT(k_w, wkT);   cvtWT(v_w, wvT);
  cvtWT(ki_w, wkiT); cvtWT(vi_w, wviT); cvtWT(o_w, woT);
  cvt(x, xb, MQ * DIMV);
  cvt(ctx, cb, (long)BATCH * NCTX * DIMV);

  // 2. Projections (f32 out, bias fused)
  gemm(xb, wqT, q_b, qf, (int)MQ);
  for (int b = 0; b < BATCH; ++b) {
    const unsigned short* txtA = cb + ((long)b * NCTX + NIMG) * DIMV;
    const unsigned short* imgA = cb + (long)b * NCTX * DIMV;
    gemm(txtA, wkT,  k_b,  ktf + (long)b * NTXT * DIMV, NTXT);
    gemm(txtA, wvT,  v_b,  vtf + (long)b * NTXT * DIMV, NTXT);
    gemm(imgA, wkiT, ki_b, kif + (long)b * NIMG * DIMV, NIMG);
    gemm(imgA, wviT, vi_b, vif + (long)b * NIMG * DIMV, NIMG);
  }

  // 3. RMS norms -> bf16 (q pre-scaled by 1/sqrt(D)); V -> per-head transpose
  k_rms_bf16<<<dim3((unsigned)MQ), 256, 0, stream>>>(
      qf, nq_s, qb16, 0.08838834764831845f, SEQ, SEQ);
  k_rms_bf16<<<dim3((unsigned)MT), 256, 0, stream>>>(ktf, nkt_s, ktb, 1.0f,
                                                     NTXT, LPAD_T);
  k_rms_bf16<<<dim3((unsigned)MI), 256, 0, stream>>>(kif, nki_s, kib, 1.0f,
                                                     NIMG, LPAD_I);
  {
    long nt = (long)BATCH * LPAD_T * DIMV;
    long ni = (long)BATCH * LPAD_I * DIMV;
    k_cvt_vT<<<dim3((unsigned)((nt + 255) / 256)), 256, 0, stream>>>(
        vtf, vTt, NTXT, LPAD_T);
    k_cvt_vT<<<dim3((unsigned)((ni + 255) / 256)), 256, 0, stream>>>(
        vif, vTi, NIMG, LPAD_I);
  }

  // 4. Two SDPA passes (independent softmaxes, summed)
  dim3 agrid(SEQ / 128, HEADS, BATCH);
  k_attn<<<agrid, 256, 0, stream>>>(qb16, ktb, vTt, attnf, attnb, SEQ, NTXT,
                                    LPAD_T, 0);
  k_attn<<<agrid, 256, 0, stream>>>(qb16, kib, vTi, attnf, attnb, SEQ, NIMG,
                                    LPAD_I, 1);

  // 5. Output projection -> f32 d_out
  gemm(attnb, woT, o_b, out, (int)MQ);
}
